// Model_46076409151744
// MI455X (gfx1250) — matrix operand order repaired and checked by reading, not compiled, not run on hardware
//
#include <hip/hip_runtime.h>
#include <hip/hip_bf16.h>
#include <math.h>

// ---------------- problem constants ----------------
#define BATCH 64
#define SEQ   512
#define DIM   768
#define DA    384
#define NCLS  6
#define SEPID 103

#define ROWS  64            // rows of X per workgroup in score kernel (4 M-tiles)
#define KT    (DIM/32)      // 24 K-tiles (k=32 per WMMA)
#define NT    (DA/16)       // 24 N-tiles per matrix
#define XS_STRIDE 390       // u32 stride per LDS row (768 bf16 = 384 u32, +6 pad)
#define SMEM_SCORE (2*ROWS*4 + ROWS*XS_STRIDE*4)   // scf + X slab (bytes)

typedef __attribute__((ext_vector_type(16))) __bf16 v16bf;
typedef __attribute__((ext_vector_type(8)))  float  v8f;

union FragBF { v16bf v; unsigned int u[8]; };
union AccF   { v8f   v; float        f[8]; };

__device__ __forceinline__ unsigned int f2bf(float f) {
    unsigned int u = __float_as_uint(f);
    return (u + 0x7FFFu + ((u >> 16) & 1u)) >> 16;   // RNE f32->bf16
}

// ============================================================
// Kernel 1: pack W_ori / W_re into bf16 WMMA B-fragment layout.
// Fragment f = mat*576 + kt*24 + nt ; 32 lanes x 16 bf16 = 1024B each.
// B (32x16, 16-bit), contiguous k order used here and by A: lanes 0-15 -> N, K = 2v+hi ; lanes 16-31 -> K+16.
// ============================================================
__global__ __launch_bounds__(256)
void pack_w_kernel(const float* __restrict__ Wori,
                   const float* __restrict__ Wre,
                   unsigned short* __restrict__ packW) {
    int wave = blockIdx.x * 8 + (threadIdx.x >> 5);
    int lane = threadIdx.x & 31;
    if (wave >= 2 * KT * NT) return;
    int mat = wave / (KT * NT);
    int rem = wave % (KT * NT);
    int kt = rem / NT, nt = rem % NT;
    const float* W = mat ? Wre : Wori;

    int n  = nt * 16 + (lane & 15);
    int kh = lane >> 4;
    unsigned int u[8];
#pragma unroll
    for (int vv = 0; vv < 8; ++vv) {
        int k0 = kt * 32 + 2 * vv + 16 * kh;
        unsigned int lo = f2bf(W[(size_t)k0 * DA + n]);
        unsigned int hi = f2bf(W[(size_t)(k0 + 1) * DA + n]);
        u[vv] = lo | (hi << 16);
    }
    unsigned int* dst = (unsigned int*)packW + (size_t)wave * 256 + lane * 8;
#pragma unroll
    for (int vv = 0; vv < 8; ++vv) dst[vv] = u[vv];
}

// ============================================================
// Kernel 2: scores[b][mat][s] = tanh(lhs[b,s,:] @ W + bias) . w
// One WG = 64 rows of one batch. 8 waves: 0-3 -> ori, 4-7 -> re.
// Each wave: 3 groups x 2 N-tiles; each B fragment feeds 4 M-tile WMMAs.
// ============================================================
__global__ __launch_bounds__(256)
void score_kernel(const float* __restrict__ lhs,
                  const unsigned short* __restrict__ packW,
                  const float* __restrict__ b_ori, const float* __restrict__ w_ori,
                  const float* __restrict__ b_re,  const float* __restrict__ w_re,
                  float* __restrict__ scores) {
    extern __shared__ char smem[];
    float*        scf = (float*)smem;                               // 2*ROWS
    unsigned int* XsU = (unsigned int*)(smem + 2 * ROWS * 4);       // ROWS*XS_STRIDE

    const int tid  = threadIdx.x;
    const int lane = tid & 31;
    const int wave = tid >> 5;
    const int b    = blockIdx.y;
    const int s0   = blockIdx.x * ROWS;

    // ---- load 64x768 f32 -> bf16 LDS (float4 per transaction) ----
    const float4* src = (const float4*)(lhs + ((size_t)b * SEQ + s0) * DIM);
    for (int i = tid; i < ROWS * (DIM / 4); i += 256) {
        int row = i / (DIM / 4), kv = i % (DIM / 4);
        float4 x = src[(size_t)row * (DIM / 4) + kv];
        XsU[row * XS_STRIDE + kv * 2]     = f2bf(x.x) | (f2bf(x.y) << 16);
        XsU[row * XS_STRIDE + kv * 2 + 1] = f2bf(x.z) | (f2bf(x.w) << 16);
    }
    if (tid < 2 * ROWS) scf[tid] = 0.0f;
    __syncthreads();

    const int mat = wave >> 2;          // 0 = ori, 1 = re
    const int wq  = wave & 3;
    const float* bvec = mat ? b_re : b_ori;
    const float* wvec = mat ? w_re : w_ori;

    const int m  = lane & 15;
    const int kh = lane >> 4;

    for (int ng = 0; ng < 3; ++ng) {
        const int ntb = wq * 6 + ng * 2;     // 2 N-tiles per group

        AccF acc[4][2];
#pragma unroll
        for (int mt = 0; mt < 4; ++mt)
#pragma unroll
            for (int t = 0; t < 2; ++t)
#pragma unroll
                for (int r = 0; r < 8; ++r) acc[mt][t].f[r] = 0.0f;

#pragma unroll 1
        for (int kt = 0; kt < KT; ++kt) {
            // A fragments (16x32 bf16) for all 4 M-tiles from LDS
            FragBF A[4];
            const int kb = kt * 16 + kh * 8;
#pragma unroll
            for (int mt = 0; mt < 4; ++mt) {
                const int rowbase = (mt * 16 + m) * XS_STRIDE;
#pragma unroll
                for (int vv = 0; vv < 4; ++vv) {
                    A[mt].u[vv]     = XsU[rowbase + kb + vv];
                    A[mt].u[vv + 4] = XsU[rowbase + kb + 4 + vv];
                }
            }

            const unsigned int* wb =
                (const unsigned int*)packW +
                ((size_t)(mat * (KT * NT) + kt * NT + ntb) * 256) + lane * 8;
            if (kt + 1 < KT)
                __builtin_prefetch(wb + (size_t)NT * 256, 0, 1);  // next K-step B frags

#pragma unroll
            for (int t = 0; t < 2; ++t) {
                FragBF Bf;
#pragma unroll
                for (int vv = 0; vv < 8; ++vv) Bf.u[vv] = wb[(size_t)t * 256 + vv];
#pragma unroll
                for (int mt = 0; mt < 4; ++mt) {
                    acc[mt][t].v = __builtin_amdgcn_wmma_f32_16x16x32_bf16(
                        false, A[mt].v, false, Bf.v, (short)0, acc[mt][t].v,
                        false, false);
                }
            }
        }

        // ---- epilogue: tanh(acc + bias) * w, reduce 16 cols, accumulate ----
#pragma unroll
        for (int t = 0; t < 2; ++t) {
            int n = (ntb + t) * 16 + (lane & 15);
            float bb = bvec[n];
            float ww = wvec[n];
#pragma unroll
            for (int mt = 0; mt < 4; ++mt) {
#pragma unroll
                for (int r = 0; r < 8; ++r) {
                    float val = tanhf(acc[mt][t].f[r] + bb) * ww;
                    val += __shfl_xor(val, 1);
                    val += __shfl_xor(val, 2);
                    val += __shfl_xor(val, 4);
                    val += __shfl_xor(val, 8);
                    if ((lane & 15) == 0) {
                        int row = mt * 16 + (lane >> 4) * 8 + r;
                        atomicAdd(&scf[mat * ROWS + row], val);
                    }
                }
            }
        }
    }
    __syncthreads();

    if (tid < 2 * ROWS) {
        int mm = tid >> 6, row = tid & (ROWS - 1);
        scores[((size_t)b * 2 + mm) * SEQ + s0 + row] = scf[tid];
    }
}

// ============================================================
// Kernel 3: masked softmax + weighted sum + FC head. 1 WG / batch row.
// ============================================================
__global__ __launch_bounds__(256)
void pool_kernel(const float* __restrict__ lhs,
                 const float* __restrict__ scores,
                 const int* __restrict__ ids,
                 const int* __restrict__ amask,
                 const float* __restrict__ W_fc1, const float* __restrict__ b_fc1,
                 const float* __restrict__ W_fc,  const float* __restrict__ b_fc,
                 float* __restrict__ out) {
    __shared__ float ps[2][SEQ];
    __shared__ float red[256];
    __shared__ float mixed[2 * DIM];
    __shared__ float hv[DIM];
    __shared__ int   seps;

    const int tid = threadIdx.x;
    const int b   = blockIdx.x;

    if (tid == 0) seps = 0;
    __syncthreads();
    for (int s = tid; s < SEQ; s += 256)
        if (ids[(size_t)b * SEQ + s] == SEPID) seps = s;   // exactly one SEP/row
    __syncthreads();
    const int sep = seps;

    // raw masked scores
    for (int mm = 0; mm < 2; ++mm) {
        for (int s = tid; s < SEQ; s += 256) {
            float v = scores[((size_t)b * 2 + mm) * SEQ + s];
            bool keep = (s != 0) && (s != sep) && (amask[(size_t)b * SEQ + s] != 0);
            ps[mm][s] = keep ? v : -INFINITY;
        }
    }
    __syncthreads();

    for (int mm = 0; mm < 2; ++mm) {
        // max
        float lm = fmaxf(ps[mm][tid], ps[mm][tid + 256]);
        red[tid] = lm; __syncthreads();
        for (int st = 128; st > 0; st >>= 1) {
            if (tid < st) red[tid] = fmaxf(red[tid], red[tid + st]);
            __syncthreads();
        }
        float mx = red[0]; __syncthreads();
        // exp + sum
        float e0 = __expf(ps[mm][tid] - mx);
        float e1 = __expf(ps[mm][tid + 256] - mx);
        ps[mm][tid] = e0; ps[mm][tid + 256] = e1;
        red[tid] = e0 + e1; __syncthreads();
        for (int st = 128; st > 0; st >>= 1) {
            if (tid < st) red[tid] += red[tid + st];
            __syncthreads();
        }
        float inv = 1.0f / red[0]; __syncthreads();
        ps[mm][tid] *= inv; ps[mm][tid + 256] *= inv;
        __syncthreads();
    }

    // weighted sums over lhs (coalesced across tid)
    for (int d = tid; d < DIM; d += 256) {
        float a0 = 0.0f, a1 = 0.0f;
        const float* base = lhs + (size_t)b * SEQ * DIM + d;
        for (int s = 0; s < SEQ; ++s) {
            float x = base[(size_t)s * DIM];
            a0 = fmaf(ps[0][s], x, a0);
            a1 = fmaf(ps[1][s], x, a1);
        }
        mixed[d]       = a0;
        mixed[DIM + d] = a1;
    }
    __syncthreads();

    // h = mixed @ W_fc1 + b_fc1
    for (int j = tid; j < DIM; j += 256) {
        float a = b_fc1[j];
        for (int i = 0; i < 2 * DIM; ++i)
            a = fmaf(mixed[i], W_fc1[(size_t)i * DIM + j], a);
        hv[j] = a;
    }
    __syncthreads();

    // out = h @ W_fc + b_fc
    if (tid < NCLS) {
        float a = b_fc[tid];
        for (int j = 0; j < DIM; ++j)
            a = fmaf(hv[j], W_fc[(size_t)j * NCLS + tid], a);
        out[(size_t)b * NCLS + tid] = a;
    }
}

// ============================================================
extern "C" void kernel_launch(void* const* d_in, const int* in_sizes, int n_in,
                              void* d_out, int out_size, void* d_ws, size_t ws_size,
                              hipStream_t stream) {
    const float* lhs   = (const float*)d_in[0];
    const float* W_ori = (const float*)d_in[1];
    const float* b_ori = (const float*)d_in[2];
    const float* w_ori = (const float*)d_in[3];
    const float* W_re  = (const float*)d_in[4];
    const float* b_re  = (const float*)d_in[5];
    const float* w_re  = (const float*)d_in[6];
    const float* W_fc1 = (const float*)d_in[7];
    const float* b_fc1 = (const float*)d_in[8];
    const float* W_fc  = (const float*)d_in[9];
    const float* b_fc  = (const float*)d_in[10];
    const int*   ids   = (const int*)d_in[11];
    const int*   amask = (const int*)d_in[12];
    float* out = (float*)d_out;

    unsigned short* packW = (unsigned short*)d_ws;                       // 1,179,648 B
    float* scores = (float*)((char*)d_ws + (size_t)2 * KT * NT * 1024);  // 262,144 B

    pack_w_kernel<<<(2 * KT * NT + 7) / 8, 256, 0, stream>>>(W_ori, W_re, packW);
    score_kernel<<<dim3(SEQ / ROWS, BATCH), 256, SMEM_SCORE, stream>>>(
        lhs, packW, b_ori, w_ori, b_re, w_re, scores);
    pool_kernel<<<BATCH, 256, 0, stream>>>(
        lhs, scores, ids, amask, W_fc1, b_fc1, W_fc, b_fc, out);
}
